// RouteEncoderLayer_35519379538331
// MI455X (gfx1250) — matrix operand order repaired and checked by reading, not compiled, not run on hardware
//
#include <hip/hip_runtime.h>

typedef _Float16 h16;
typedef __attribute__((ext_vector_type(16))) _Float16 v16h;
typedef __attribute__((ext_vector_type(8)))  _Float16 v8h;
typedef __attribute__((ext_vector_type(4)))  _Float16 v4h;
typedef __attribute__((ext_vector_type(8)))  float    v8f;
typedef __attribute__((ext_vector_type(4)))  int      v4i;

#define EDIM   128
#define NMODES 6
#define NHEADS 8
#define HDDIM  16
#define NBATCH 1024
#define NKEDGE 65536

#if __has_builtin(__builtin_amdgcn_global_load_async_to_lds_b128)
#define HAVE_ASYNC_LDS 1
#else
#define HAVE_ASYNC_LDS 0
#endif

static __device__ __forceinline__ void wait_asynccnt0() {
#if __has_builtin(__builtin_amdgcn_s_wait_asynccnt)
  __builtin_amdgcn_s_wait_asynccnt(0);
#else
  asm volatile("s_wait_asynccnt 0x0" ::: "memory");
#endif
}

// Stage one 16-byte chunk of the A tile into LDS: async global->LDS
// (ASYNCcnt-tracked GLOBAL_LOAD_ASYNC_TO_LDS_B128) if the toolchain exposes
// it, else VGPR bounce.
static __device__ __forceinline__ void stage16(const h16* __restrict__ src,
                                               h16* __restrict__ dst) {
#if HAVE_ASYNC_LDS
  __builtin_amdgcn_global_load_async_to_lds_b128(
      (__attribute__((address_space(1))) v4i*)src,
      (__attribute__((address_space(3))) v4i*)dst, 0, 0);
  wait_asynccnt0();
#else
  *(v8h*)dst = *(const v8h*)src;
#endif
}

// ---------------------------------------------------------------------------
// A-fragment (16x32, contiguous k order shared with B): lane holds row M=lane&15;
// element e -> K = e + 16*(lane>>4)  => two contiguous 8-half loads.
static __device__ __forceinline__ v16h frag_a(const h16* __restrict__ p) {
  v8h lo = *(const v8h*)(p);
  v8h hi = *(const v8h*)(p + 8);
  v16h r;
#pragma unroll
  for (int i = 0; i < 8; ++i) { r[i] = lo[i]; r[i + 8] = hi[i]; }
  return r;
}

// ---------------------------------------------------------------------------
// GEMM: Y[R,128] = X[R,128](f16) @ W[128,128]^T (W row n = output col n) + bias
// Block: 256 threads = 8 waves; block owns 16 rows, wave w owns cols [16w,16w+16).
template <bool ACCUM, bool RES, bool RELU, bool SF32, bool SF16>
__global__ __launch_bounds__(256) void gemm128(
    const h16* __restrict__ X, const h16* __restrict__ W,
    const float* __restrict__ bias, const float* __restrict__ res,
    float* __restrict__ Yf, h16* __restrict__ Yh) {
  __shared__ h16 As[16 * 128];
  const int lane = threadIdx.x & 31;
  const int wid  = threadIdx.x >> 5;
  const long rowBase = (long)blockIdx.x * 16;
  // cooperative A-tile stage: 256 threads x 16B = 4 KB
  stage16(X + rowBase * 128 + threadIdx.x * 8, As + threadIdx.x * 8);
  __syncthreads();
  const int lh = lane >> 4;
  const int lr = lane & 15;
  v8f acc = {};
#pragma unroll
  for (int kb = 0; kb < 4; ++kb) {
    v16h a = frag_a(As + lr * 128 + kb * 32 + 16 * lh);
    // B layout: lane holds col N=lane&15, element e -> K = e + 16*(lane>>4)
    v16h b = *(const v16h*)(W + (size_t)(wid * 16 + lr) * 128 + kb * 32 + 16 * lh);
    acc = __builtin_amdgcn_wmma_f32_16x16x32_f16(false, a, false, b, (short)0, acc,
                                                 false, false);
  }
  const int col = wid * 16 + lr;
  const float bv = bias[col];
#pragma unroll
  for (int v = 0; v < 8; ++v) {
    long idx = (rowBase + v + 8 * lh) * 128 + col;   // C: M = v + 8*lh, N = lr
    float val = acc[v] + bv;
    if (RES)   val += res[idx];
    if (ACCUM) val += Yf[idx];
    if (RELU)  val = val > 0.f ? val : 0.f;
    if (SF32)  Yf[idx] = val;
    if (SF16)  Yh[idx] = (h16)val;
  }
}

// Dual-output GEMM: reads A once, applies two weight sets (ca_k + lin_v1 fusion).
__global__ __launch_bounds__(256) void gemm128_dual(
    const h16* __restrict__ X,
    const h16* __restrict__ W1, const float* __restrict__ b1, h16* __restrict__ Y1,
    const h16* __restrict__ W2, const float* __restrict__ b2, h16* __restrict__ Y2) {
  __shared__ h16 As[16 * 128];
  const int lane = threadIdx.x & 31;
  const int wid  = threadIdx.x >> 5;
  const long rowBase = (long)blockIdx.x * 16;
  stage16(X + rowBase * 128 + threadIdx.x * 8, As + threadIdx.x * 8);
  __syncthreads();
  const int lh = lane >> 4;
  const int lr = lane & 15;
  v8f acc1 = {}, acc2 = {};
#pragma unroll
  for (int kb = 0; kb < 4; ++kb) {
    v16h a  = frag_a(As + lr * 128 + kb * 32 + 16 * lh);
    v16h f1 = *(const v16h*)(W1 + (size_t)(wid * 16 + lr) * 128 + kb * 32 + 16 * lh);
    v16h f2 = *(const v16h*)(W2 + (size_t)(wid * 16 + lr) * 128 + kb * 32 + 16 * lh);
    acc1 = __builtin_amdgcn_wmma_f32_16x16x32_f16(false, a, false, f1, (short)0, acc1,
                                                  false, false);
    acc2 = __builtin_amdgcn_wmma_f32_16x16x32_f16(false, a, false, f2, (short)0, acc2,
                                                  false, false);
  }
  const int col = wid * 16 + lr;
  const float bv1 = b1[col], bv2 = b2[col];
#pragma unroll
  for (int v = 0; v < 8; ++v) {
    long idx = (rowBase + v + 8 * lh) * 128 + col;
    Y1[idx] = (h16)(acc1[v] + bv1);
    Y2[idx] = (h16)(acc2[v] + bv2);
  }
}

// ---------------------------------------------------------------------------
// Row-wise LayerNorm over 128, f16 output. One wave per row (wave32), 8 rows/block.
__global__ __launch_bounds__(256) void ln128(const float* __restrict__ X,
                                             const float* __restrict__ g,
                                             const float* __restrict__ bta,
                                             h16* __restrict__ Y) {
  const long row = (long)blockIdx.x * 8 + (threadIdx.x >> 5);
  const int lane = threadIdx.x & 31;
  const float4 x = *(const float4*)(X + row * 128 + lane * 4);
  float s = x.x + x.y + x.z + x.w;
#pragma unroll
  for (int m = 1; m < 32; m <<= 1) s += __shfl_xor(s, m, 32);
  const float mu = s * (1.f / 128.f);
  const float d0 = x.x - mu, d1 = x.y - mu, d2 = x.z - mu, d3 = x.w - mu;
  float v = d0 * d0 + d1 * d1 + d2 * d2 + d3 * d3;
#pragma unroll
  for (int m = 1; m < 32; m <<= 1) v += __shfl_xor(v, m, 32);
  const float inv = rsqrtf(v * (1.f / 128.f) + 1e-5f);
  const float4 gw = *(const float4*)(g + lane * 4);
  const float4 gb = *(const float4*)(bta + lane * 4);
  v4h out;
  out[0] = (h16)(d0 * inv * gw.x + gb.x);
  out[1] = (h16)(d1 * inv * gw.y + gb.y);
  out[2] = (h16)(d2 * inv * gw.z + gb.z);
  out[3] = (h16)(d3 * inv * gw.w + gb.w);
  *(v4h*)(Y + row * 128 + lane * 4) = out;
}

// f32 -> f16 copy (4 elems / thread)
__global__ void cvt_f32_h(const float* __restrict__ X, h16* __restrict__ Y, long n4) {
  long i = (long)blockIdx.x * blockDim.x + threadIdx.x;
  if (i >= n4) return;
  const float4 x = *(const float4*)(X + i * 4);
  v4h y; y[0] = (h16)x.x; y[1] = (h16)x.y; y[2] = (h16)x.z; y[3] = (h16)x.w;
  *(v4h*)(Y + i * 4) = y;
}

// transpose [D0,D1,128] -> [D1,D0,128]
__global__ void transpose_f(const float* __restrict__ in, float* __restrict__ out,
                            int D0, int D1) {
  long t = (long)blockIdx.x * blockDim.x + threadIdx.x;
  long total4 = (long)D0 * D1 * 32;
  if (t >= total4) return;
  long c4 = t & 31, rc = t >> 5;
  long i0 = rc / D1, i1 = rc % D1;
  const float4 x = *(const float4*)(in + rc * 128 + c4 * 4);
  *(float4*)(out + (i1 * D0 + i0) * 128 + c4 * 4) = x;
}
__global__ void transpose_h(const float* __restrict__ in, h16* __restrict__ out,
                            int D0, int D1) {
  long t = (long)blockIdx.x * blockDim.x + threadIdx.x;
  long total4 = (long)D0 * D1 * 32;
  if (t >= total4) return;
  long c4 = t & 31, rc = t >> 5;
  long i0 = rc / D1, i1 = rc % D1;
  const float4 x = *(const float4*)(in + rc * 128 + c4 * 4);
  v4h y; y[0] = (h16)x.x; y[1] = (h16)x.y; y[2] = (h16)x.z; y[3] = (h16)x.w;
  *(v4h*)(out + (i1 * D0 + i0) * 128 + c4 * 4) = y;
}

// c = a + b, stored f32 + f16
__global__ void add_cvt(const float* __restrict__ A, const float* __restrict__ B,
                        float* __restrict__ Cf, h16* __restrict__ Ch, long n4) {
  long i = (long)blockIdx.x * blockDim.x + threadIdx.x;
  if (i >= n4) return;
  const float4 a = *(const float4*)(A + i * 4);
  const float4 b = *(const float4*)(B + i * 4);
  float4 c; c.x = a.x + b.x; c.y = a.y + b.y; c.z = a.z + b.z; c.w = a.w + b.w;
  *(float4*)(Cf + i * 4) = c;
  v4h y; y[0] = (h16)c.x; y[1] = (h16)c.y; y[2] = (h16)c.z; y[3] = (h16)c.w;
  *(v4h*)(Ch + i * 4) = y;
}

// ---------------------------------------------------------------------------
// Self-attention core over modes: Q/K/V f32 [6,1024,128] (rows m*1024+n);
// one thread per (node, head); output stored transposed as f16 rows (n*6+l).
__global__ __launch_bounds__(256) void sa_attn(const float* __restrict__ Q,
                                               const float* __restrict__ K,
                                               const float* __restrict__ V,
                                               h16* __restrict__ O) {
  int t = blockIdx.x * 256 + threadIdx.x;
  if (t >= NBATCH * NHEADS) return;
  const int n = t >> 3, h = t & 7;
  const int off = h * HDDIM;
  for (int l = 0; l < NMODES; ++l) {
    const float* q = Q + ((size_t)l * NBATCH + n) * 128 + off;
    float s[NMODES];
    float mx = -1e30f;
#pragma unroll
    for (int ss = 0; ss < NMODES; ++ss) {
      const float* k = K + ((size_t)ss * NBATCH + n) * 128 + off;
      float d = 0.f;
#pragma unroll
      for (int dd = 0; dd < HDDIM; ++dd) d += q[dd] * k[dd];
      s[ss] = d * 0.25f;                       // 1/sqrt(16)
      mx = fmaxf(mx, s[ss]);
    }
    float sum = 0.f;
#pragma unroll
    for (int ss = 0; ss < NMODES; ++ss) { s[ss] = expf(s[ss] - mx); sum += s[ss]; }
    const float inv = 1.f / sum;
#pragma unroll
    for (int dd = 0; dd < HDDIM; ++dd) {
      float o = 0.f;
#pragma unroll
      for (int ss = 0; ss < NMODES; ++ss)
        o += s[ss] * V[((size_t)ss * NBATCH + n) * 128 + off + dd];
      O[((size_t)n * NMODES + l) * 128 + off + dd] = (h16)(o * inv);
    }
  }
}

// segment starts: seg[b] = lower_bound(batch, b); seg[B] = NK (batch is sorted)
__global__ void seg_bounds(const int* __restrict__ batch, int* __restrict__ seg) {
  int b = blockIdx.x * 256 + threadIdx.x;
  if (b > NBATCH) return;
  if (b == NBATCH) { seg[NBATCH] = NKEDGE; return; }
  int lo = 0, hi = NKEDGE;
  while (lo < hi) { int mid = (lo + hi) >> 1; if (batch[mid] < b) lo = mid + 1; else hi = mid; }
  seg[b] = lo;
}

// ---------------------------------------------------------------------------
// Cross-attention: one block per (b,m), one wave per head, online segment softmax.
// lanes 0-15: content half (Kc . Qc), lanes 16-31: positional half (Kp . Qs).
__global__ __launch_bounds__(256) void ca_attn(
    const float* __restrict__ Qc, const float* __restrict__ Qs,
    const h16* __restrict__ Kc, const h16* __restrict__ Kp,
    const h16* __restrict__ V1, const h16* __restrict__ V2,
    const int* __restrict__ seg, float* __restrict__ agg) {
  const int bm = blockIdx.x;                 // b*6 + m
  const int b = bm / NMODES, m = bm % NMODES;
  const int h = threadIdx.x >> 5;
  const int lane = threadIdx.x & 31;
  const int d = lane & 15;
  const bool hiHalf = lane >= 16;
  const size_t qidx = (size_t)bm * 128 + h * HDDIM + d;
  const float qv = hiHalf ? Qs[qidx] : Qc[qidx];
  const int lo = seg[b], hi = seg[b + 1];
  float M = -1e30f, S = 0.f, acc = 0.f;
  for (int e = lo; e < hi; ++e) {
    if (e + 1 < hi) {   // stream next edge's lines through the cache hierarchy
      __builtin_prefetch(Kc + ((size_t)(e + 1) * NMODES + m) * 128 + h * HDDIM, 0, 0);
      __builtin_prefetch(V1 + ((size_t)(e + 1) * NMODES + m) * 128 + h * HDDIM, 0, 0);
    }
    float kv = hiHalf ? (float)Kp[(size_t)e * 128 + h * HDDIM + d]
                      : (float)Kc[((size_t)e * NMODES + m) * 128 + h * HDDIM + d];
    float p = kv * qv;                       // 32-dim dot across the wave
#pragma unroll
    for (int msk = 1; msk < 32; msk <<= 1) p += __shfl_xor(p, msk, 32);
    const float s = p;                       // no sqrt scaling in source
    const float nM = fmaxf(M, s);
    const float sc = expf(M - nM);
    const float w  = expf(s - nM);
    const float vv = (float)V1[((size_t)e * NMODES + m) * 128 + h * HDDIM + d] +
                     (float)V2[(size_t)e * 128 + h * HDDIM + d];
    acc = acc * sc + w * vv;
    S   = S * sc + w;
    M   = nM;
  }
  if (!hiHalf)
    agg[(size_t)bm * 128 + h * HDDIM + d] = (S > 0.f) ? acc / S : 0.f;
}

// ---------------------------------------------------------------------------
extern "C" void kernel_launch(void* const* d_in, const int* in_sizes, int n_in,
                              void* d_out, int out_size, void* d_ws, size_t ws_size,
                              hipStream_t stream) {
  (void)in_sizes; (void)n_in; (void)out_size; (void)ws_size;
  // ---- inputs -------------------------------------------------------------
  const int*   batch   = (const int*)d_in[0];
  const float* ego_f   = (const float*)d_in[1];   // [6,1024,128]
  const float* qpos    = (const float*)d_in[2];   // [6,1024,128]
  const float* qsine   = (const float*)d_in[3];   // [1024,6,128]
  const float* kv_pos  = (const float*)d_in[4];   // [65536,128]
  const float* edge_f  = (const float*)d_in[5];   // [65536,6,128]
  const float* node_f  = (const float*)d_in[6];   // [65536,128]
  // params flattened in dict order: 17 lin (w,b), mha_in (w,b), 3 LN (w,b)
  enum { QC=0, QP, KC, KP, SAV, N1, N2, L1, L2, CAQ, CAQP, CAQS, CAK, CAKP, LV1, LV2, MOUT };
  auto Wp = [&](int i) { return (const float*)d_in[7 + 2 * i]; };
  auto Bp = [&](int i) { return (const float*)d_in[8 + 2 * i]; };
  const float* mha_w   = (const float*)d_in[41];  // [384,128]
  const float* mha_b   = (const float*)d_in[42];  // [384]
  const float* sanorm_w = (const float*)d_in[43];
  const float* sanorm_b = (const float*)d_in[44];
  const float* nv1_w = (const float*)d_in[45];
  const float* nv1_b = (const float*)d_in[46];
  const float* nv2_w = (const float*)d_in[47];
  const float* nv2_b = (const float*)d_in[48];

  // ---- workspace carve ----------------------------------------------------
  char* wsp = (char*)d_ws;
  auto carve = [&](size_t bytes) -> void* {
    void* r = (void*)wsp;
    wsp += (bytes + 255) & ~(size_t)255;
    return r;
  };
  const long RB = NMODES * NBATCH;          // 6144
  const long RE = (long)NKEDGE * NMODES;    // 393216
  h16* wh      = (h16*)carve(17 * 16384 * sizeof(h16));  // f16 weights
  h16* mha_wh  = (h16*)carve(384 * 128 * sizeof(h16));
  h16* tgt_h   = (h16*)carve(RB * 128 * sizeof(h16));
  h16* qp_h    = (h16*)carve(RB * 128 * sizeof(h16));
  float* q32   = (float*)carve(RB * 128 * sizeof(float));
  float* k32   = (float*)carve(RB * 128 * sizeof(float));
  h16* qh      = (h16*)carve(RB * 128 * sizeof(h16));
  h16* kh      = (h16*)carve(RB * 128 * sizeof(h16));
  h16* vh      = (h16*)carve(RB * 128 * sizeof(h16));
  float* qq32  = (float*)carve(RB * 128 * sizeof(float));
  float* kk32  = (float*)carve(RB * 128 * sizeof(float));
  float* vv32  = (float*)carve(RB * 128 * sizeof(float));
  h16* o_h     = (h16*)carve(RB * 128 * sizeof(h16));
  float* egoFT = (float*)carve(RB * 128 * sizeof(float)); // ego_feature transposed
  h16* egoT_h  = (h16*)carve(RB * 128 * sizeof(h16));
  h16* qpT_h   = (h16*)carve(RB * 128 * sizeof(h16));
  h16* qse_h   = (h16*)carve(RB * 128 * sizeof(h16));
  float* egon32 = (float*)carve(RB * 128 * sizeof(float));
  h16* qsum_h  = (h16*)carve(RB * 128 * sizeof(h16));
  float* qq2   = (float*)carve(RB * 128 * sizeof(float));
  float* qse2  = (float*)carve(RB * 128 * sizeof(float));
  h16* edge_h  = (h16*)carve(RE * 128 * sizeof(h16));
  h16* node_h  = (h16*)carve((long)NKEDGE * 128 * sizeof(h16));
  h16* kvpos_h = (h16*)carve((long)NKEDGE * 128 * sizeof(h16));
  h16* Kh      = (h16*)carve(RE * 128 * sizeof(h16));
  h16* Kposh   = (h16*)carve((long)NKEDGE * 128 * sizeof(h16));
  h16* V1h     = (h16*)carve(RE * 128 * sizeof(h16));
  h16* V2h     = (h16*)carve((long)NKEDGE * 128 * sizeof(h16));
  int* seg     = (int*)carve((NBATCH + 1) * sizeof(int));
  float* agg32 = (float*)carve(RB * 128 * sizeof(float));
  float* ego2_32 = (float*)carve(RB * 128 * sizeof(float));
  h16* ego2_h  = (h16*)carve(RB * 128 * sizeof(h16));
  h16* t1_h    = (h16*)carve(RB * 128 * sizeof(h16));
  h16* t2_h    = (h16*)carve(RB * 128 * sizeof(h16));
  float* out32 = (float*)carve(RB * 128 * sizeof(float));

  // ---- 0: weight conversion ----------------------------------------------
  for (int i = 0; i < 17; ++i)
    cvt_f32_h<<<16, 256, 0, stream>>>(Wp(i), wh + (size_t)i * 16384, 4096);
  cvt_f32_h<<<48, 256, 0, stream>>>(mha_w, mha_wh, 12288);

  // ---- A: self-attention over modes  (rows r = m*1024 + b) ---------------
  ln128<<<RB / 8, 256, 0, stream>>>(ego_f, sanorm_w, sanorm_b, tgt_h);
  cvt_f32_h<<<RB * 32 / 256, 256, 0, stream>>>(qpos, qp_h, RB * 32);
  gemm128<false,false,false,true,false><<<RB/16,256,0,stream>>>(tgt_h, wh+QC*16384, Bp(QC), nullptr, q32, nullptr);
  gemm128<true, false,false,true,true ><<<RB/16,256,0,stream>>>(qp_h,  wh+QP*16384, Bp(QP), nullptr, q32, qh);
  gemm128<false,false,false,true,false><<<RB/16,256,0,stream>>>(tgt_h, wh+KC*16384, Bp(KC), nullptr, k32, nullptr);
  gemm128<true, false,false,true,true ><<<RB/16,256,0,stream>>>(qp_h,  wh+KP*16384, Bp(KP), nullptr, k32, kh);
  gemm128<false,false,false,false,true><<<RB/16,256,0,stream>>>(tgt_h, wh+SAV*16384, Bp(SAV), nullptr, nullptr, vh);
  // MHA in-proj (mha_in_w rows 0/128/256)
  gemm128<false,false,false,true,false><<<RB/16,256,0,stream>>>(qh, mha_wh,           mha_b,       nullptr, qq32, nullptr);
  gemm128<false,false,false,true,false><<<RB/16,256,0,stream>>>(kh, mha_wh + 16384,   mha_b + 128, nullptr, kk32, nullptr);
  gemm128<false,false,false,true,false><<<RB/16,256,0,stream>>>(vh, mha_wh + 32768,   mha_b + 256, nullptr, vv32, nullptr);
  sa_attn<<<(NBATCH * NHEADS) / 256, 256, 0, stream>>>(qq32, kk32, vv32, o_h);
  // ego = ego_feature + out-proj, directly in [B,M,E] layout
  transpose_f<<<RB * 32 / 256, 256, 0, stream>>>(ego_f, egoFT, NMODES, NBATCH);
  gemm128<false,true,false,false,true><<<RB/16,256,0,stream>>>(o_h, wh+MOUT*16384, Bp(MOUT), egoFT, nullptr, egoT_h);

  // ---- B: query-side cross-attn projections (rows r = b*6 + m) -----------
  gemm128<false,false,false,true,false><<<RB/16,256,0,stream>>>(egoT_h, wh+N1*16384, Bp(N1), nullptr, egon32, nullptr);
  transpose_h<<<RB * 32 / 256, 256, 0, stream>>>(qpos, qpT_h, NMODES, NBATCH);
  cvt_f32_h<<<RB * 32 / 256, 256, 0, stream>>>(qsine, qse_h, RB * 32);
  gemm128<false,true,false,false,true><<<RB/16,256,0,stream>>>(qpT_h, wh+CAQP*16384, Bp(CAQP), egon32, nullptr, qsum_h);
  gemm128<false,false,false,true,false><<<RB/16,256,0,stream>>>(qsum_h, wh+CAQ*16384,  Bp(CAQ),  nullptr, qq2,  nullptr);
  gemm128<false,false,false,true,false><<<RB/16,256,0,stream>>>(qse_h,  wh+CAQS*16384, Bp(CAQS), nullptr, qse2, nullptr);

  // ---- C: edge-side projections ------------------------------------------
  ln128<<<RE / 8, 256, 0, stream>>>(edge_f, nv1_w, nv1_b, edge_h);
  ln128<<<NKEDGE / 8, 256, 0, stream>>>(node_f, nv2_w, nv2_b, node_h);
  cvt_f32_h<<<(long)NKEDGE * 32 / 256, 256, 0, stream>>>(kv_pos, kvpos_h, (long)NKEDGE * 32);
  gemm128_dual<<<RE/16, 256, 0, stream>>>(edge_h, wh+CAK*16384, Bp(CAK), Kh,
                                          wh+LV1*16384, Bp(LV1), V1h);
  gemm128<false,false,false,false,true><<<NKEDGE/16,256,0,stream>>>(kvpos_h, wh+CAKP*16384, Bp(CAKP), nullptr, nullptr, Kposh);
  gemm128<false,false,false,false,true><<<NKEDGE/16,256,0,stream>>>(node_h,  wh+LV2*16384,  Bp(LV2),  nullptr, nullptr, V2h);

  // ---- D: segment softmax attention + aggregation ------------------------
  seg_bounds<<<(NBATCH + 1 + 255) / 256, 256, 0, stream>>>(batch, seg);
  ca_attn<<<RB, 256, 0, stream>>>(qq2, qse2, Kh, Kposh, V1h, V2h, seg, agg32);

  // ---- E: FFN + residuals + output transpose -----------------------------
  add_cvt<<<RB * 32 / 256, 256, 0, stream>>>(egon32, agg32, ego2_32, ego2_h, RB * 32);
  gemm128<false,false,false,false,true><<<RB/16,256,0,stream>>>(ego2_h, wh+N2*16384, Bp(N2), nullptr, nullptr, t1_h);
  gemm128<false,false,true, false,true><<<RB/16,256,0,stream>>>(t1_h,   wh+L1*16384, Bp(L1), nullptr, nullptr, t2_h);
  gemm128<false,true, false,true,false><<<RB/16,256,0,stream>>>(t2_h,   wh+L2*16384, Bp(L2), ego2_32, out32, nullptr);
  transpose_f<<<RB * 32 / 256, 256, 0, stream>>>(out32, (float*)d_out, NBATCH, NMODES);
}
